// Net_3195455668714
// MI455X (gfx1250) — matrix operand order repaired and checked by reading, not compiled, not run on hardware
//
#include <hip/hip_runtime.h>
#include <hip/hip_bf16.h>
#include <stdint.h>

typedef __bf16 bf16_t;
typedef __attribute__((ext_vector_type(16))) __bf16 v16bf;
typedef __attribute__((ext_vector_type(8)))  float  v8f;

#define DEVINL __device__ __forceinline__

// ---------------- geometry ----------------
static constexpr int H  = 160, W  = 160;   // image
static constexpr int HP = 167, WP = 167;   // backbone feature map (H + 2p - 1, p=4)
static constexpr int NPIX = HP * WP;       // 27889
static constexpr int NOUT = H * W;         // 25600
static constexpr int C  = 32;              // channels
static constexpr int R  = 192;             // projection dim
static constexpr int KP = 2048;            // patch dim = 32*8*8 (K permuted: (dy*8+dx)*32+c)
static constexpr int NS = 1024;            // padded SVD sample rows (1000 real)

// ---------------- helpers ----------------
DEVINL uint32_t pcg(uint32_t x){ x = x*747796405u + 2891336453u;
  uint32_t w = ((x >> ((x>>28)+4u)) ^ x) * 277803737u; return (w>>22)^w; }
DEVINL float u01(uint32_t h){ return (float)(h >> 8) * (1.0f/16777216.0f); }

DEVINL v16bf zero_frag(){
  union { v16bf v; uint32_t u[8]; } z;
#pragma unroll
  for (int i=0;i<8;i++) z.u[i]=0u;
  return z.v;
}
// A fragment (16x32 bf16): lane l (h=l/16, m=l%16) holds K = h*8..h*8+7 and 16+h*8..+7
DEVINL v16bf load_a_frag(const bf16_t* row32, int h){
  union { v16bf v; uint4 q[2]; } u;
  u.q[0] = *reinterpret_cast<const uint4*>(row32 + h*8);
  u.q[1] = *reinterpret_cast<const uint4*>(row32 + 16 + h*8);
  return u.v;
}
// B fragment (32x16 bf16 row-major tile, row stride ld): lane l (h=l/16, n=l%16) holds column n, K = h*8..h*8+7 and 16+h*8..+7
DEVINL v16bf load_b_frag(const bf16_t* col, int ld, int h){
  v16bf v;
#pragma unroll
  for (int s = 0; s < 8; ++s){
    v[s]     = col[(size_t)(h*8 + s)*ld];
    v[8 + s] = col[(size_t)(16 + h*8 + s)*ld];
  }
  return v;
}
DEVINL v8f wmma_bf16(v16bf a, v16bf b, v8f c){
  return __builtin_amdgcn_wmma_f32_16x16x32_bf16(false, a, false, b, (short)0, c, false, false);
}
DEVINL v8f zero_acc(){ v8f z = {0.f,0.f,0.f,0.f,0.f,0.f,0.f,0.f}; return z; }

// Stage a 32xR bf16 chunk (12 KB) from global into LDS via the CDNA5 async
// global->LDS DMA path (ASYNCcnt). 128 threads x 6 b128 transfers = 768 x 16B.
// grow = first row of the chunk (row stride R). Ends with wave wait + barrier.
DEVINL void stage_b_chunk(const bf16_t* __restrict__ grow, bf16_t* sm, int tid){
#pragma unroll
  for (int q = 0; q < 6; ++q){
    const int e   = tid + 128*q;          // 0..767
    const int row = e / 24, seg = e % 24; // 24 x 16B segments per 192-elem row
    uint32_t ldst = (uint32_t)(uintptr_t)(sm + row*R + seg*8);
    uint64_t ga   = (uint64_t)(uintptr_t)(grow + (size_t)row*R + seg*8);
    asm volatile("global_load_async_to_lds_b128 %0, %1, off"
                 :: "v"(ldst), "v"(ga) : "memory");
  }
  asm volatile("s_wait_asynccnt 0x0" ::: "memory");
  __syncthreads();
}

// ---------------- param prep ----------------
__global__ void k_fuse_bn(const float* __restrict__ conv0_b, const float* __restrict__ bn0_g,
                          const float* __restrict__ bn0_b, const float* __restrict__ bn0_m,
                          const float* __restrict__ bn0_v,
                          const float* __restrict__ b1, const float* __restrict__ g1,
                          const float* __restrict__ be1, const float* __restrict__ m1,
                          const float* __restrict__ v1,
                          const float* __restrict__ b2, const float* __restrict__ g2,
                          const float* __restrict__ be2, const float* __restrict__ m2,
                          const float* __restrict__ v2,
                          const float* __restrict__ conv1_b, float* __restrict__ sb)
{
  int tid = blockIdx.x*256 + threadIdx.x;
  if (tid >= 18*32) return;
  int e = tid >> 5, ch = tid & 31;
  float scale, bias;
  if (e == 0){
    scale = bn0_g[ch] * rsqrtf(bn0_v[ch] + 1e-5f);
    bias  = (conv0_b[ch] - bn0_m[ch]) * scale + bn0_b[ch];
  } else if (e <= 16){
    int k = (e-1) >> 1, idx = k*32 + ch;
    bool second = ((e-1) & 1) != 0;
    const float* bb = second ? b2 : b1;
    const float* gg = second ? g2 : g1;
    const float* be_= second ? be2: be1;
    const float* mm = second ? m2 : m1;
    const float* vv = second ? v2 : v1;
    scale = gg[idx] * rsqrtf(vv[idx] + 1e-5f);
    bias  = (bb[idx] - mm[idx]) * scale + be_[idx];
  } else { scale = 1.0f; bias = conv1_b[ch]; }
  sb[e*64 + ch]      = scale;
  sb[e*64 + 32 + ch] = bias;
}

// pack 17 3x3 conv layers into B layout [L][kg=(ky*3+kx)*32+c][o] bf16
__global__ void k_pack_w(const float* __restrict__ w1, const float* __restrict__ w2,
                         const float* __restrict__ wlast, bf16_t* __restrict__ wb)
{
  int tid = blockIdx.x*256 + threadIdx.x;
  if (tid >= 17*288*32) return;
  int o  = tid & 31;
  int kg = (tid >> 5) % 288;
  int L  = tid / (288*32);
  int tap = kg >> 5, c = kg & 31;
  int ky = tap / 3, kx = tap % 3;
  float w;
  if (L < 16){
    int k = L >> 1;
    const float* src = (L & 1) ? w2 : w1;
    w = src[(((k*32 + o)*32 + c)*3 + ky)*3 + kx];
  } else {
    w = wlast[((o*32 + c)*3 + ky)*3 + kx];
  }
  wb[tid] = (bf16_t)w;
}

// ---------------- backbone ----------------
__global__ void k_pad(const float* __restrict__ img, float* __restrict__ imgp){
  int tid = blockIdx.x*256 + threadIdx.x;
  if (tid >= NPIX*3) return;
  int c = tid % 3, pix = tid / 3;
  int y = pix / WP, x = pix % WP;
  int y0 = y - 4, x0 = x - 4;
  float v = 0.f;
  if ((unsigned)y0 < (unsigned)H && (unsigned)x0 < (unsigned)W) v = img[(c*H + y0)*W + x0];
  imgp[pix*3 + c] = v;
}

__global__ void k_conv0(const float* __restrict__ imgp, const float* __restrict__ w,
                        const float* __restrict__ sb, bf16_t* __restrict__ out){
  int tid = blockIdx.x*256 + threadIdx.x;
  if (tid >= NPIX*C) return;
  int ch = tid & 31, pix = tid >> 5;
  int y = pix / WP, x = pix % WP;
  float acc = 0.f;
  for (int c = 0; c < 3; ++c)
    for (int ky = 0; ky < 5; ++ky){
      int yy = y + ky - 2;
      if ((unsigned)yy >= (unsigned)HP) continue;
      for (int kx = 0; kx < 5; ++kx){
        int xx = x + kx - 2;
        if ((unsigned)xx >= (unsigned)WP) continue;
        acc += imgp[(yy*WP+xx)*3 + c] * w[((ch*3+c)*5+ky)*5+kx];
      }
    }
  float v = acc * sb[ch] + sb[32+ch];
  out[pix*C + ch] = (bf16_t)fmaxf(v, 0.f);
}

// implicit-GEMM 3x3 conv, 32->32, bf16 HWC in/out, f32 accumulate via WMMA.
// block = 4 waves, wave computes 16 pixels x 32 channels.
__global__ void k_conv3x3(const bf16_t* __restrict__ in, const bf16_t* __restrict__ wb,
                          const float* __restrict__ sb, const bf16_t* __restrict__ resid,
                          int relu, bf16_t* __restrict__ out)
{
  const int wave = threadIdx.x >> 5, lane = threadIdx.x & 31;
  const int hh = lane >> 4, low = lane & 15;
  const int mbase = (blockIdx.x * 4 + wave) * 16;
  int m = mbase + low; if (m > NPIX-1) m = NPIX-1;
  const int y = m / WP, x = m % WP;
  v8f acc0 = zero_acc(), acc1 = zero_acc();
#pragma unroll
  for (int t = 0; t < 9; ++t) {
    const int yy = y + t/3 - 1, xx = x + t%3 - 1;
    v16bf a = ((unsigned)yy < (unsigned)HP && (unsigned)xx < (unsigned)WP)
                ? load_a_frag(in + (yy*WP + xx)*C, hh) : zero_frag();
    const bf16_t* brow = wb + (t*32)*C + low;
    acc0 = wmma_bf16(a, load_b_frag(brow, C, hh), acc0);
    acc1 = wmma_bf16(a, load_b_frag(brow + 16, C, hh), acc1);
  }
#pragma unroll
  for (int r = 0; r < 8; ++r) {
    const int mo = mbase + r + 8*hh;
    if (mo < NPIX) {
      float v0 = acc0[r] * sb[low]    + sb[32 + low];
      float v1 = acc1[r] * sb[16+low] + sb[48 + low];
      if (resid) { v0 += (float)resid[mo*C + low]; v1 += (float)resid[mo*C + 16 + low]; }
      if (relu)  { v0 = fmaxf(v0, 0.f); v1 = fmaxf(v1, 0.f); }
      out[mo*C + low]      = (bf16_t)v0;
      out[mo*C + 16 + low] = (bf16_t)v1;
    }
  }
}

// ---------------- approximate SVD (Gram + block power iteration) ----------------
__global__ void k_sample(const bf16_t* __restrict__ feat, bf16_t* __restrict__ xs){
  int tid = blockIdx.x*256 + threadIdx.x;
  if (tid >= NS*64) return;
  int t = tid & 63, i = tid >> 6;
  uint4* dst = reinterpret_cast<uint4*>(xs + (size_t)i*KP + t*32);
  if (i < 1000){
    uint32_t pix = pcg((uint32_t)i*2654435761u ^ 0x9E3779B9u) % (uint32_t)NOUT;
    int iy = (int)pix / W, ix = (int)pix % W;
    int dy = t >> 3, dx = t & 7;
    const uint4* src = reinterpret_cast<const uint4*>(feat + ((iy+dy)*WP + (ix+dx))*C);
#pragma unroll
    for (int q=0;q<4;q++) dst[q] = src[q];
  } else {
    uint4 z; z.x=0u; z.y=0u; z.z=0u; z.w=0u;
#pragma unroll
    for (int q=0;q<4;q++) dst[q] = z;
  }
}

__global__ void k_transpose(const bf16_t* __restrict__ xs, bf16_t* __restrict__ xst){
  int tid = blockIdx.x*256 + threadIdx.x;
  if (tid >= NS*KP) return;
  int i  = tid & (NS-1);
  int kg = tid >> 10;
  xst[(size_t)kg*NS + i] = xs[(size_t)i*KP + kg];
}

// G = Xs^T Xs  (2048x2048, K=1024), output bf16
__global__ void k_gram(const bf16_t* __restrict__ xst, const bf16_t* __restrict__ xs,
                       bf16_t* __restrict__ g)
{
  const int wave = threadIdx.x>>5, lane = threadIdx.x&31;
  const int hh = lane>>4, low = lane&15;
  const int mbase = blockIdx.x * 16;
  const int n0 = (blockIdx.y*4 + wave) * 16;
  v8f acc = zero_acc();
  for (int t = 0; t < 32; ++t){
    v16bf a = load_a_frag(xst + (size_t)(mbase+low)*NS + t*32, hh);
    v16bf b = load_b_frag(xs + (size_t)(t*32)*KP + n0 + low, KP, hh);
    acc = wmma_bf16(a, b, acc);
  }
#pragma unroll
  for (int r=0;r<8;r++)
    g[(size_t)(mbase + r + 8*hh)*KP + n0 + low] = (bf16_t)acc[r];
}

__global__ void k_initV(bf16_t* __restrict__ v){
  int tid = blockIdx.x*256 + threadIdx.x;
  if (tid >= KP*R) return;
  v[tid] = (bf16_t)(u01(pcg((uint32_t)tid ^ 0x6A09E667u))*2.f - 1.f);
}

// Vn = G @ Vbf  (2048x2048 @ 2048x192); B chunk staged in LDS via async DMA
__global__ void __launch_bounds__(128) k_gv(const bf16_t* __restrict__ g,
                                            const bf16_t* __restrict__ vin,
                                            float* __restrict__ vout)
{
  __shared__ __align__(16) bf16_t bsm[32*R];   // 12 KB
  const int tid = threadIdx.x;
  const int wave = tid>>5, lane = tid&31;
  const int hh = lane>>4, low = lane&15;
  const int mbase = (blockIdx.x*4 + wave)*16;
  v8f acc[12];
#pragma unroll
  for (int q=0;q<12;q++) acc[q] = zero_acc();
  for (int t = 0; t < 64; ++t){
    stage_b_chunk(vin + (size_t)t*32*R, bsm, tid);
    v16bf a = load_a_frag(g + (size_t)(mbase+low)*KP + t*32, hh);
#pragma unroll
    for (int q=0;q<12;q++) acc[q] = wmma_bf16(a, load_b_frag(&bsm[q*16 + low], R, hh), acc[q]);
    __syncthreads();
  }
#pragma unroll
  for (int q=0;q<12;q++)
#pragma unroll
    for (int r=0;r<8;r++)
      vout[(size_t)(mbase + r + 8*hh)*R + q*16 + low] = acc[q][r];
}

__global__ void k_colnorm(const float* __restrict__ vin, bf16_t* __restrict__ vbf,
                          float* __restrict__ norms)
{
  __shared__ float red[256];
  const int r = blockIdx.x;
  float s = 0.f;
  for (int k = threadIdx.x; k < KP; k += 256){ float v = vin[(size_t)k*R + r]; s += v*v; }
  red[threadIdx.x] = s; __syncthreads();
  for (int st = 128; st > 0; st >>= 1){
    if (threadIdx.x < st) red[threadIdx.x] += red[threadIdx.x + st];
    __syncthreads();
  }
  const float nrm = sqrtf(red[0]);
  const float inv = 1.0f / fmaxf(nrm, 1e-20f);
  if (threadIdx.x == 0) norms[r] = nrm;
  for (int k = threadIdx.x; k < KP; k += 256)
    vbf[(size_t)k*R + r] = (bf16_t)(vin[(size_t)k*R + r] * inv);
}

__global__ void k_limit(const float* __restrict__ norms, float* __restrict__ mask){
  if (threadIdx.x == 0 && blockIdx.x == 0){
    float tot = 0.f;
    for (int r = 0; r < R; ++r){ float s = norms[r]; tot += s*s; }
    float cs = 0.f; int limit = 0;
    for (int r = 0; r < R; ++r){ float s = norms[r]; cs += s*s; if (cs < 0.995f*tot) limit++; }
    limit += 1;
    for (int r = 0; r < R; ++r) mask[r] = (r < limit) ? 1.f : 0.f;
  }
}

__global__ void k_packproj(const bf16_t* __restrict__ vbf, const float* __restrict__ mask,
                           bf16_t* __restrict__ pbw){
  int tid = blockIdx.x*256 + threadIdx.x;
  if (tid >= KP*R) return;
  int r = tid % R;
  pbw[tid] = (bf16_t)((float)vbf[tid] * mask[r]);
}

// ---------------- patch projection: pa = patches(feat) @ proj ----------------
// implicit patch extraction: K chunk t -> tap (dy,dx), 32 contiguous channels.
// B chunk (32x192, 12 KB) staged into LDS once per block via async DMA and
// shared by all 4 waves; next A row prefetched one chunk ahead.
__global__ void __launch_bounds__(128) k_patchproj(const bf16_t* __restrict__ feat,
                                                   const bf16_t* __restrict__ pbw,
                                                   float* __restrict__ outp)
{
  __shared__ __align__(16) bf16_t bsm[32*R];   // 12 KB
  const int tid = threadIdx.x;
  const int wave = tid>>5, lane = tid&31;
  const int hh = lane>>4, low = lane&15;
  const int mbase = (blockIdx.x*4 + wave)*16;
  int m = mbase + low; if (m > NOUT-1) m = NOUT-1;
  const int i = m / W, j = m % W;
  v8f acc[12];
#pragma unroll
  for (int q=0;q<12;q++) acc[q] = zero_acc();
  for (int t = 0; t < 64; ++t){
    stage_b_chunk(pbw + (size_t)t*32*R, bsm, tid);
    const int dy = t >> 3, dx = t & 7;
    v16bf a = load_a_frag(feat + ((i+dy)*WP + (j+dx))*C, hh);
    if (t < 63){
      const int dy2 = (t+1) >> 3, dx2 = (t+1) & 7;
      __builtin_prefetch(feat + ((i+dy2)*WP + (j+dx2))*C, 0, 1);
    }
#pragma unroll
    for (int q=0;q<12;q++) acc[q] = wmma_bf16(a, load_b_frag(&bsm[q*16 + low], R, hh), acc[q]);
    __syncthreads();
  }
#pragma unroll
  for (int q=0;q<12;q++)
#pragma unroll
    for (int r=0;r<8;r++){
      const int mo = mbase + r + 8*hh;
      if (mo < NOUT) outp[(size_t)mo*R + q*16 + low] = acc[q][r];
    }
}

// ---------------- PatchMatch ----------------
__global__ void k_rownorm(const float* __restrict__ p, float* __restrict__ n2){
  const int wave = threadIdx.x>>5, lane = threadIdx.x&31;
  const int row = blockIdx.x*8 + wave;
  if (row >= NOUT) return;
  float s = 0.f;
#pragma unroll
  for (int q=0;q<6;q++){ float v = p[(size_t)row*R + lane + 32*q]; s += v*v; }
#pragma unroll
  for (int st=16;st>0;st>>=1) s += __shfl_xor(s, st, 32);
  if (lane == 0) n2[row] = s;
}

__global__ void k_pm_init(const float* __restrict__ pa, const float* __restrict__ pb,
                          const float* __restrict__ na2, const float* __restrict__ nb2,
                          int* __restrict__ map0, float* __restrict__ dist0)
{
  const int wave = threadIdx.x>>5, lane = threadIdx.x&31;
  const int n = blockIdx.x*8 + wave;
  if (n >= NOUT) return;
  const int m = (int)(pcg((uint32_t)n*2654435761u ^ 0x13579BDFu) % (uint32_t)NOUT);
  float partial = 0.f;
#pragma unroll
  for (int q=0;q<6;q++) partial += pa[(size_t)n*R + lane + 32*q] * pb[(size_t)m*R + lane + 32*q];
#pragma unroll
  for (int st=16;st>0;st>>=1) partial += __shfl_xor(partial, st, 32);
  if (lane == 0){ map0[n] = m; dist0[n] = na2[n] + nb2[m] - 2.f*partial; }
}

__global__ void k_pm_iter(const float* __restrict__ pa, const float* __restrict__ pb,
                          const float* __restrict__ na2, const float* __restrict__ nb2,
                          const int* __restrict__ mapIn, const float* __restrict__ distIn,
                          int* __restrict__ mapOut, float* __restrict__ distOut, int it)
{
  const int wave = threadIdx.x>>5, lane = threadIdx.x&31;
  const int n = blockIdx.x*8 + wave;
  if (n >= NOUT) return;
  const int curM = mapIn[n];
  const float curD = distIn[n];
  if (lane == 0){ mapOut[n] = curM; distOut[n] = curD; }
  if (!(curD > 0.01f)) return;

  float av[6];
#pragma unroll
  for (int q=0;q<6;q++) av[q] = pa[(size_t)n*R + lane + 32*q];

  int nbrs[8];
  nbrs[0] = n - ((n >= 1) ? 1 : 0);
  nbrs[1] = n + ((n < NOUT-1) ? 1 : 0);
  nbrs[2] = n - ((n >= W) ? W : 0);
  nbrs[3] = n + ((n < NOUT-W) ? W : 0);
  nbrs[4] = n - ((n >= 8*W) ? 8*W : 0);
  nbrs[5] = n + ((n < NOUT-8*W) ? 8*W : 0);
  nbrs[6] = n - ((n >= 8) ? 8 : 0);
  nbrs[7] = n + ((n < NOUT-8) ? 8 : 0);

  const int o = curM - n;
  const int tdiv = (int)truncf((float)o / (float)W);
  int omod = o % W; if (omod < 0) omod += W;

  float bestD = curD; int bestM = curM;
  for (int c = 0; c < 14; ++c){
    int b;
    if (c < 8){
      int nb = nbrs[c];
      b = n + (mapIn[nb] - nb);
    } else {
      int kk = c - 8;
      float bound = (float)W * exp2f(-(float)(kk+1));   // 80,40,20,10,5,2.5
      uint32_t hx = pcg((uint32_t)n*2654435761u ^ (uint32_t)((it+1)*0x85EBCA6B) ^ (uint32_t)(kk*0x9E3779B1));
      uint32_t hy = pcg(hx ^ 0x5F5F5F5Fu);
      int rx = (int)((u01(hx)*2.f - 1.f) * bound);
      int ry = (int)((u01(hy)*2.f - 1.f) * bound);
      b = n + ((tdiv + rx)*W + omod + ry);
    }
    b = (b >= 0) ? b : 0;
    b = (b < NOUT) ? b : 0;
    float partial = 0.f;
#pragma unroll
    for (int q=0;q<6;q++) partial += av[q] * pb[(size_t)b*R + lane + 32*q];
#pragma unroll
    for (int st=16;st>0;st>>=1) partial += __shfl_xor(partial, st, 32);
    float d = na2[n] + nb2[b] - 2.f*partial;
    if (d < bestD){ bestD = d; bestM = b; }
  }
  if (lane == 0){ mapOut[n] = bestM; distOut[n] = bestD; }
}

__global__ void k_final(const int* __restrict__ mapv, float* __restrict__ out){
  int n = blockIdx.x*256 + threadIdx.x;
  if (n >= NOUT) return;
  int m = mapv[n];
  int my = m / W, mx = m % W;
  int gy = n / W, gx = n % W;
  out[2*n]     = (float)(my - gy);
  out[2*n + 1] = (float)(mx - gx);
  out[2*NOUT + 2*n]     = (float)my;
  out[2*NOUT + 2*n + 1] = (float)mx;
}

// ---------------- host ----------------
extern "C" void kernel_launch(void* const* d_in, const int* in_sizes, int n_in,
                              void* d_out, int out_size, void* d_ws, size_t ws_size,
                              hipStream_t stream)
{
  (void)in_sizes; (void)n_in; (void)out_size; (void)ws_size;
  const float* image_a = (const float*)d_in[0];
  const float* image_b = (const float*)d_in[1];
  const float* conv0_w = (const float*)d_in[2];
  const float* conv0_b = (const float*)d_in[3];
  const float* bn0_g   = (const float*)d_in[4];
  const float* bn0_b   = (const float*)d_in[5];
  const float* bn0_m   = (const float*)d_in[6];
  const float* bn0_v   = (const float*)d_in[7];
  const float* blk_w1  = (const float*)d_in[8];
  const float* blk_b1  = (const float*)d_in[9];
  const float* blk_g1  = (const float*)d_in[10];
  const float* blk_be1 = (const float*)d_in[11];
  const float* blk_m1  = (const float*)d_in[12];
  const float* blk_v1  = (const float*)d_in[13];
  const float* blk_w2  = (const float*)d_in[14];
  const float* blk_b2  = (const float*)d_in[15];
  const float* blk_g2  = (const float*)d_in[16];
  const float* blk_be2 = (const float*)d_in[17];
  const float* blk_m2  = (const float*)d_in[18];
  const float* blk_v2  = (const float*)d_in[19];
  const float* conv1_w = (const float*)d_in[20];
  const float* conv1_b = (const float*)d_in[21];
  // d_in[22]=patch_size(8), d_in[23]=iterations(5): fixed per setup_inputs

  size_t off = 0;
  auto carve = [&](size_t bytes)->void*{
    void* p = (void*)((char*)d_ws + off);
    off += (bytes + 255) & ~(size_t)255;
    return p;
  };
  float*  imgP  = (float*) carve((size_t)NPIX*3*4);
  bf16_t* actX  = (bf16_t*)carve((size_t)NPIX*C*2);
  bf16_t* actH  = (bf16_t*)carve((size_t)NPIX*C*2);
  bf16_t* actT  = (bf16_t*)carve((size_t)NPIX*C*2);
  bf16_t* featA = (bf16_t*)carve((size_t)NPIX*C*2);
  bf16_t* featB = (bf16_t*)carve((size_t)NPIX*C*2);
  bf16_t* Wb    = (bf16_t*)carve((size_t)17*288*32*2);
  float*  sbAll = (float*) carve((size_t)18*64*4);
  bf16_t* Xs    = (bf16_t*)carve((size_t)NS*KP*2);
  bf16_t* XsT   = (bf16_t*)carve((size_t)KP*NS*2);
  bf16_t* G     = (bf16_t*)carve((size_t)KP*KP*2);
  float*  Vn    = (float*) carve((size_t)KP*R*4);
  bf16_t* Vbf   = (bf16_t*)carve((size_t)KP*R*2);
  float*  norms = (float*) carve((size_t)R*4);
  float*  maskp = (float*) carve((size_t)R*4);
  bf16_t* Pb    = (bf16_t*)carve((size_t)KP*R*2);
  float*  pa    = (float*) carve((size_t)NOUT*R*4);
  float*  pb    = (float*) carve((size_t)NOUT*R*4);
  float*  na2   = (float*) carve((size_t)NOUT*4);
  float*  nb2   = (float*) carve((size_t)NOUT*4);
  int*    mapA  = (int*)   carve((size_t)NOUT*4);
  int*    mapB  = (int*)   carve((size_t)NOUT*4);
  float*  distA = (float*) carve((size_t)NOUT*4);
  float*  distB = (float*) carve((size_t)NOUT*4);

  // ---- param prep ----
  k_fuse_bn<<<3, 256, 0, stream>>>(conv0_b, bn0_g, bn0_b, bn0_m, bn0_v,
                                   blk_b1, blk_g1, blk_be1, blk_m1, blk_v1,
                                   blk_b2, blk_g2, blk_be2, blk_m2, blk_v2,
                                   conv1_b, sbAll);
  k_pack_w<<<(17*288*32 + 255)/256, 256, 0, stream>>>(blk_w1, blk_w2, conv1_w, Wb);

  // ---- backbone, both images ----
  const float* imgs[2]  = {image_a, image_b};
  bf16_t*      feats[2] = {featA, featB};
  for (int im = 0; im < 2; ++im){
    k_pad  <<<(NPIX*3 + 255)/256, 256, 0, stream>>>(imgs[im], imgP);
    k_conv0<<<(NPIX*C + 255)/256, 256, 0, stream>>>(imgP, conv0_w, sbAll, actX);
    bf16_t* cur = actX; bf16_t* tmp = actT;
    for (int k = 0; k < 8; ++k){
      k_conv3x3<<<436, 128, 0, stream>>>(cur,  Wb + (size_t)(2*k)*288*32,   sbAll + (1+2*k)*64, nullptr, 1, actH);
      k_conv3x3<<<436, 128, 0, stream>>>(actH, Wb + (size_t)(2*k+1)*288*32, sbAll + (2+2*k)*64, cur,     1, tmp);
      bf16_t* sw = cur; cur = tmp; tmp = sw;
    }
    k_conv3x3<<<436, 128, 0, stream>>>(cur, Wb + (size_t)16*288*32, sbAll + 17*64, nullptr, 0, feats[im]);
  }

  // ---- approximate SVD projection ----
  k_sample   <<<(NS*64 + 255)/256, 256, 0, stream>>>(featA, Xs);
  k_transpose<<<(NS*KP + 255)/256, 256, 0, stream>>>(Xs, XsT);
  k_gram     <<<dim3(128, 32), 128, 0, stream>>>(XsT, Xs, G);
  k_initV    <<<(KP*R + 255)/256, 256, 0, stream>>>(Vbf);
  for (int it = 0; it < 8; ++it){
    k_gv     <<<32, 128, 0, stream>>>(G, Vbf, Vn);
    k_colnorm<<<R, 256, 0, stream>>>(Vn, Vbf, norms);
  }
  k_limit   <<<1, 32, 0, stream>>>(norms, maskp);
  k_packproj<<<(KP*R + 255)/256, 256, 0, stream>>>(Vbf, maskp, Pb);

  // ---- patch projection GEMMs (dominant FLOPs) ----
  k_patchproj<<<400, 128, 0, stream>>>(featA, Pb, pa);
  k_patchproj<<<400, 128, 0, stream>>>(featB, Pb, pb);
  k_rownorm  <<<NOUT/8, 256, 0, stream>>>(pa, na2);
  k_rownorm  <<<NOUT/8, 256, 0, stream>>>(pb, nb2);

  // ---- PatchMatch ----
  k_pm_init<<<NOUT/8, 256, 0, stream>>>(pa, pb, na2, nb2, mapA, distA);
  int* mi = mapA; float* di = distA; int* mo = mapB; float* dd = distB;
  for (int it = 0; it < 5; ++it){
    k_pm_iter<<<NOUT/8, 256, 0, stream>>>(pa, pb, na2, nb2, mi, di, mo, dd, it);
    int* t1 = mi; mi = mo; mo = t1;
    float* t2 = di; di = dd; dd = t2;
  }
  k_final<<<(NOUT + 255)/256, 256, 0, stream>>>(mi, (float*)d_out);
}
